// PointTransformer_34144990003376
// MI455X (gfx1250) — matrix operand order repaired and checked by reading, not compiled, not run on hardware
//
#include <hip/hip_runtime.h>
#include <cmath>

// ---------------------------------------------------------------------------
// Point Transformer forward for gfx1250 (MI455X).
// All dense GEMMs run through v_wmma_f32_16x16x32_f16 (f16 A/B staged in LDS,
// f32 accumulate).  The interior-tile loop contains only: 8x global_load_b128
// + packed cvt + 4x ds_store_b128 (staging), barrier, 8x ds_load_b128
// (fragments), 4x v_wmma.  Guarded staging lives in a separate loop taken
// only by edge row-blocks and the tiny K=3/6 GEMMs.  FPS / kNN / softmax /
// LayerNorm are wave32 VALU kernels.  Workspace ~295 MB (kNN scratch aliased
// with the MLP ping-pong buffers).
// ---------------------------------------------------------------------------

#define NPTS   32768
#define BATCH  2
#define NGRP   512      // groups per batch
#define GS     32       // group size
#define DMODEL 384
#define SEQ    513
#define NHEAD  6
#define HDIM   64
#define NLAYER 12

typedef __attribute__((ext_vector_type(16))) _Float16 v16h;
typedef __attribute__((ext_vector_type(8)))  _Float16 v8h;
typedef __attribute__((ext_vector_type(8)))  float    v8f;

// ============================ WMMA GEMM ====================================
// C[M,N] = act( (A[M,K] @ W[N,K]^T + bias) * (scale*scaleC) + bias2 ) + residual
// act: 0=none 1=relu 2=gelu(exact).  Any of bias/scale/bias2/residual may be null.
#define TM 64
#define TN 64
#define TK 32
#define LDSTR 40   // halves per LDS row (80B, 16B aligned, padded vs 32)

// Fast staging: 16 consecutive f32 -> 16 f16 in LDS (4x b128 load, 2x b128 store)
__device__ __forceinline__ void stage16_fast(const float* __restrict__ src,
                                             _Float16* __restrict__ dst)
{
    const float4* s4 = (const float4*)src;
    const float4 a = s4[0], b = s4[1], c = s4[2], d = s4[3];
    v8h h0, h1;
    h0[0] = (_Float16)a.x; h0[1] = (_Float16)a.y; h0[2] = (_Float16)a.z; h0[3] = (_Float16)a.w;
    h0[4] = (_Float16)b.x; h0[5] = (_Float16)b.y; h0[6] = (_Float16)b.z; h0[7] = (_Float16)b.w;
    h1[0] = (_Float16)c.x; h1[1] = (_Float16)c.y; h1[2] = (_Float16)c.z; h1[3] = (_Float16)c.w;
    h1[4] = (_Float16)d.x; h1[5] = (_Float16)d.y; h1[6] = (_Float16)d.z; h1[7] = (_Float16)d.w;
    *(v8h*)dst       = h0;
    *(v8h*)(dst + 8) = h1;
}

// Guarded staging (edge tiles / tiny K): counted loop, zero-fill the rest.
__device__ __forceinline__ void stage16_guard(const float* __restrict__ src,
                                              _Float16* __restrict__ dst, int valid)
{
    int j = 0;
    for (; j < valid; ++j) dst[j] = (_Float16)src[j];
    for (; j < 16; ++j)    dst[j] = (_Float16)0.0f;
}

// Fragment loads (CDNA5 16-bit A/B layouts) + 4x WMMA accumulate.
__device__ __forceinline__ void mma_step(const _Float16* __restrict__ lA,
                                         const _Float16* __restrict__ lB,
                                         int wm, int wn, int lr, int hi,
                                         v8f acc[2][2])
{
    v16h aF[2], bF[2];
#pragma unroll
    for (int t = 0; t < 2; ++t) {
        // A: row = wm+t*16+lr ; halves 0..7 at K=(hi?8:0)+j, 8..15 at +16
        const _Float16* ab = &lA[(wm + t * 16 + lr) * LDSTR + (hi << 3)];
        v8h a0 = *(const v8h*)(ab);
        v8h a1 = *(const v8h*)(ab + 16);
        v16h av;
#pragma unroll
        for (int j = 0; j < 8; ++j) { av[j] = a0[j]; av[j + 8] = a1[j]; }
        aF[t] = av;
        // B: col = wn+t*16+lr ; halves 0..7 at K=(hi?8:0)+j, 8..15 at +16
        const _Float16* bb = &lB[(wn + t * 16 + lr) * LDSTR + (hi << 3)];
        v8h b0 = *(const v8h*)(bb);
        v8h b1 = *(const v8h*)(bb + 16);
        v16h bv;
#pragma unroll
        for (int j = 0; j < 8; ++j) { bv[j] = b0[j]; bv[j + 8] = b1[j]; }
        bF[t] = bv;
    }
#pragma unroll
    for (int i = 0; i < 2; ++i)
#pragma unroll
        for (int j = 0; j < 2; ++j)
            acc[i][j] = __builtin_amdgcn_wmma_f32_16x16x32_f16(
                false, aF[i], false, bF[j], (short)0, acc[i][j], false, false);
}

__launch_bounds__(128)
__global__ void gemm_wmma_kernel(const float* __restrict__ A,
                                 const float* __restrict__ W,
                                 const float* __restrict__ bias,
                                 const float* __restrict__ scale, float scaleC,
                                 const float* __restrict__ bias2,
                                 const float* __restrict__ residual,
                                 float* __restrict__ C,
                                 int M, int N, int K, int act)
{
    __shared__ _Float16 lA[TM * LDSTR];
    __shared__ _Float16 lB[TN * LDSTR];

    const int tid  = threadIdx.x;
    const int lane = tid & 31;
    const int wave = tid >> 5;          // 4 waves, 2x2 arrangement
    const int wm   = (wave >> 1) * 32;  // wave row offset in tile
    const int wn   = (wave & 1) * 32;   // wave col offset in tile
    const int rowBlk = blockIdx.y * TM;
    const int colBlk = blockIdx.x * TN;

    const int lr = lane & 15;
    const int hi = lane >> 4;           // 0 for lanes 0-15, 1 for 16-31

    // staging assignment: thread covers row r = tid>>1, k-segment ks = (tid&1)*16
    const int sr = tid >> 1;
    const int sk = (tid & 1) * 16;
    const int grA = rowBlk + sr;
    const int gcW = colBlk + sr;
    _Float16* dstA = &lA[sr * LDSTR + sk];
    _Float16* dstW = &lB[sr * LDSTR + sk];

    v8f acc[2][2] = {};

    // block-uniform, loop-invariant fast-path predicate
    const bool allFast = ((K & (TK - 1)) == 0) &&
                         (rowBlk + TM <= M) && (colBlk + TN <= N);

    if (allFast) {
        const float* srcA = A + (size_t)grA * K + sk;
        const float* srcW = W + (size_t)gcW * K + sk;
        for (int kk = 0; kk < K; kk += TK) {
            stage16_fast(srcA + kk, dstA);
            stage16_fast(srcW + kk, dstW);
            __syncthreads();
            mma_step(lA, lB, wm, wn, lr, hi, acc);
            __syncthreads();
        }
    } else {
        for (int kk = 0; kk < K; kk += TK) {
            const int krem = K - kk - sk;
            const int vA = (grA < M) ? max(0, min(16, krem)) : 0;
            const int vW = (gcW < N) ? max(0, min(16, krem)) : 0;
            stage16_guard(A + (size_t)grA * K + kk + sk, dstA, vA);
            stage16_guard(W + (size_t)gcW * K + kk + sk, dstW, vW);
            __syncthreads();
            mma_step(lA, lB, wm, wn, lr, hi, acc);
            __syncthreads();
        }
    }

    // ---- epilogue: C layout  M = v + (lane<16?0:8), N = lane&15 ----
#pragma unroll
    for (int i = 0; i < 2; ++i) {
#pragma unroll
        for (int j = 0; j < 2; ++j) {
            const int col = colBlk + wn + j * 16 + lr;
            if (col >= N) continue;
            const float bv = bias  ? bias[col]           : 0.0f;
            const float sv = scale ? scale[col] * scaleC : 1.0f;
            const float b2 = bias2 ? bias2[col]          : 0.0f;
            const int baseR = rowBlk + wm + i * 16 + hi * 8;
#pragma unroll
            for (int v = 0; v < 8; ++v) {
                const int row = baseR + v;
                if (row >= M) continue;
                float y = acc[i][j][v] + bv;
                if (scale) y = y * sv + b2;
                if (act == 1)      y = fmaxf(y, 0.0f);
                else if (act == 2) y = 0.5f * y * (1.0f + erff(y * 0.70710678118654752f));
                if (residual) y += residual[(size_t)row * N + col];
                C[(size_t)row * N + col] = y;
            }
        }
    }
}

// ============================ FPS ==========================================
// One block per batch.  Per-thread distance cache lives in registers
// (32 points per thread, index = tid + t*1024).  Argmax tie -> lowest index.
__launch_bounds__(1024)
__global__ void fps_kernel(const float* __restrict__ pts, float* __restrict__ center)
{
    const int b = blockIdx.x;
    const int tid = threadIdx.x;
    __shared__ float smax[1024];
    __shared__ int   sidx[1024];
    __shared__ float cx, cy, cz;
    __shared__ int   cur;

    float dist[32];
#pragma unroll
    for (int t = 0; t < 32; ++t) dist[t] = 1e10f;
    if (tid == 0) cur = 0;
    __syncthreads();

    for (int it = 0; it < NGRP; ++it) {
        if (tid == 0) {
            const float* p = pts + ((size_t)b * NPTS + cur) * 6;
            cx = p[0]; cy = p[1]; cz = p[2];
            float* c = center + ((size_t)b * NGRP + it) * 3;
            c[0] = cx; c[1] = cy; c[2] = cz;
        }
        __syncthreads();

        float best = -1e30f; int bi = 0;
#pragma unroll
        for (int t = 0; t < 32; ++t) {
            const int i = tid + t * 1024;
            const float* p = pts + ((size_t)b * NPTS + i) * 6;
            const float dx = p[0] - cx, dy = p[1] - cy, dz = p[2] - cz;
            const float d = dx * dx + dy * dy + dz * dz;
            const float nd = fminf(dist[t], d);
            dist[t] = nd;
            if (nd > best) { best = nd; bi = i; }   // ascending i: first max kept
        }
        smax[tid] = best; sidx[tid] = bi;
        __syncthreads();
        for (int s = 512; s > 0; s >>= 1) {
            if (tid < s) {
                const float v2 = smax[tid + s]; const int i2 = sidx[tid + s];
                if (v2 > smax[tid] || (v2 == smax[tid] && i2 < sidx[tid])) {
                    smax[tid] = v2; sidx[tid] = i2;
                }
            }
            __syncthreads();
        }
        if (tid == 0) cur = sidx[0];
        __syncthreads();
    }
}

// ============================ kNN (top-32 smallest) ========================
// One block per group; per-thread-owned global scratch slice (i % 256 == tid),
// 32 rounds of block argmin with lowest-index tie-break (== lax.top_k order).
__launch_bounds__(256)
__global__ void knn_kernel(const float* __restrict__ pts,
                           const float* __restrict__ center,
                           int* __restrict__ idx, float* __restrict__ scratch)
{
    const int g = blockIdx.x;          // 0..1023
    const int b = g >> 9;
    const int tid = threadIdx.x;
    float* scr = scratch + (size_t)g * NPTS;
    __shared__ float rv[256];
    __shared__ int   ri[256];

    const float cx = center[g * 3 + 0];
    const float cy = center[g * 3 + 1];
    const float cz = center[g * 3 + 2];

    for (int i = tid; i < NPTS; i += 256) {
        const float* p = pts + ((size_t)b * NPTS + i) * 6;
        const float dx = p[0] - cx, dy = p[1] - cy, dz = p[2] - cz;
        scr[i] = dx * dx + dy * dy + dz * dz;
    }
    __syncthreads();

    for (int sel = 0; sel < GS; ++sel) {
        float best = 1e30f; int bi = NPTS;
        for (int i = tid; i < NPTS; i += 256) {
            const float d = scr[i];
            if (d < best) { best = d; bi = i; }  // ascending i: first min kept
        }
        rv[tid] = best; ri[tid] = bi;
        __syncthreads();
        for (int s = 128; s > 0; s >>= 1) {
            if (tid < s) {
                if (rv[tid + s] < rv[tid] ||
                    (rv[tid + s] == rv[tid] && ri[tid + s] < ri[tid])) {
                    rv[tid] = rv[tid + s]; ri[tid] = ri[tid + s];
                }
            }
            __syncthreads();
        }
        if (tid == 0) idx[g * GS + sel] = ri[0];
        const int w = ri[0];
        __syncthreads();
        if ((w & 255) == tid) scr[w] = 1e30f;   // owner thread masks winner
        __syncthreads();
    }
}

// ======================= neighborhood features =============================
__global__ void build_f0_kernel(const float* __restrict__ pts,
                                const float* __restrict__ center,
                                const int* __restrict__ idx,
                                float* __restrict__ f0)
{
    const int e = blockIdx.x * 256 + threadIdx.x;
    const int total = BATCH * NGRP * GS * 6;
    if (e >= total) return;
    const int c = e % 6;
    const int r = e / 6;
    const int i = r % GS;
    const int g = r / GS;
    const int b = g >> 9;
    const int n = idx[g * GS + i];
    float v = pts[((size_t)b * NPTS + n) * 6 + c];
    if (c < 3) v -= center[g * 3 + c];
    f0[e] = v;
}

// fg = max over group, f2c = concat([fg broadcast, f2])
__launch_bounds__(256)
__global__ void concat_max_kernel(const float* __restrict__ f2, float* __restrict__ f2c)
{
    const int g = blockIdx.x;
    const int tid = threadIdx.x;     // 256 = channels of f2
    __shared__ float fg[256];
    float m = -1e30f;
    for (int i = 0; i < GS; ++i)
        m = fmaxf(m, f2[((size_t)g * GS + i) * 256 + tid]);
    fg[tid] = m;
    __syncthreads();
    for (int i = 0; i < GS; ++i) {
        float* dst = &f2c[((size_t)g * GS + i) * 512];
        dst[tid]       = fg[tid];
        dst[256 + tid] = f2[((size_t)g * GS + i) * 256 + tid];
    }
}

__launch_bounds__(256)
__global__ void featmax_kernel(const float* __restrict__ f4, float* __restrict__ feat)
{
    const int g = blockIdx.x;
    const int tid = threadIdx.x;
    float m = -1e30f;
    for (int i = 0; i < GS; ++i)
        m = fmaxf(m, f4[((size_t)g * GS + i) * 256 + tid]);
    feat[(size_t)g * 256 + tid] = m;
}

// tok [B,512,D] + cls [D]  ->  out [B,513,D]
__global__ void assemble_kernel(const float* __restrict__ tok,
                                const float* __restrict__ cls,
                                float* __restrict__ out)
{
    const int e = blockIdx.x * 256 + threadIdx.x;
    const int total = BATCH * SEQ * DMODEL;
    if (e >= total) return;
    const int d = e % DMODEL;
    const int r = e / DMODEL;
    const int s = r % SEQ;
    const int b = r / SEQ;
    out[e] = (s == 0) ? cls[d] : tok[((size_t)b * NGRP + (s - 1)) * DMODEL + d];
}

// ============================ LayerNorm ====================================
// One block (128 threads) per row of 384.  Optionally xi = x + pos stored.
__launch_bounds__(128)
__global__ void ln_kernel(const float* __restrict__ x, const float* __restrict__ pos,
                          float* __restrict__ xi,
                          const float* __restrict__ g, const float* __restrict__ bt,
                          float* __restrict__ out)
{
    const int row = blockIdx.x;
    const int tid = threadIdx.x;
    __shared__ float red[128];
    const size_t base = (size_t)row * DMODEL;

    float vals[3];
    float lsum = 0.0f;
#pragma unroll
    for (int t = 0; t < 3; ++t) {
        const int d = tid + t * 128;
        float v = x[base + d];
        if (pos) v += pos[base + d];
        if (xi)  xi[base + d] = v;
        vals[t] = v;
        lsum += v;
    }
    red[tid] = lsum; __syncthreads();
    for (int s = 64; s > 0; s >>= 1) { if (tid < s) red[tid] += red[tid + s]; __syncthreads(); }
    const float mean = red[0] / (float)DMODEL;
    __syncthreads();

    float lvar = 0.0f;
#pragma unroll
    for (int t = 0; t < 3; ++t) { const float d = vals[t] - mean; lvar += d * d; }
    red[tid] = lvar; __syncthreads();
    for (int s = 64; s > 0; s >>= 1) { if (tid < s) red[tid] += red[tid + s]; __syncthreads(); }
    const float inv = rsqrtf(red[0] / (float)DMODEL + 1e-5f);

#pragma unroll
    for (int t = 0; t < 3; ++t) {
        const int d = tid + t * 128;
        out[base + d] = g[d] * (vals[t] - mean) * inv + bt[d];
    }
}

// ============================ Attention ====================================
// grid (S, H, B), 128 threads.  One query row per block; S=513 scores in LDS.
__launch_bounds__(128)
__global__ void attn_kernel(const float* __restrict__ qkv, float* __restrict__ o)
{
    const int qi = blockIdx.x;
    const int h  = blockIdx.y;
    const int b  = blockIdx.z;
    const int tid = threadIdx.x;
    __shared__ float qv[HDIM];
    __shared__ float sc[SEQ];
    __shared__ float red[128];

    const size_t rowq = ((size_t)(b * SEQ + qi)) * (3 * DMODEL);
    if (tid < HDIM) qv[tid] = qkv[rowq + h * HDIM + tid];
    __syncthreads();

    for (int k = tid; k < SEQ; k += 128) {
        const float* kp = &qkv[((size_t)(b * SEQ + k)) * (3 * DMODEL) + DMODEL + h * HDIM];
        float s = 0.0f;
#pragma unroll
        for (int d = 0; d < HDIM; ++d) s += qv[d] * kp[d];
        sc[k] = s * 0.125f;                         // 1/sqrt(64)
    }
    __syncthreads();

    float m = -1e30f;
    for (int k = tid; k < SEQ; k += 128) m = fmaxf(m, sc[k]);
    red[tid] = m; __syncthreads();
    for (int s = 64; s > 0; s >>= 1) { if (tid < s) red[tid] = fmaxf(red[tid], red[tid + s]); __syncthreads(); }
    m = red[0];
    __syncthreads();

    float lsum = 0.0f;
    for (int k = tid; k < SEQ; k += 128) { const float e = expf(sc[k] - m); sc[k] = e; lsum += e; }
    red[tid] = lsum; __syncthreads();
    for (int s = 64; s > 0; s >>= 1) { if (tid < s) red[tid] += red[tid + s]; __syncthreads(); }
    const float inv = 1.0f / red[0];
    __syncthreads();

    if (tid < HDIM) {
        float a = 0.0f;
        for (int k = 0; k < SEQ; ++k)
            a += sc[k] * qkv[((size_t)(b * SEQ + k)) * (3 * DMODEL) + 2 * DMODEL + h * HDIM + tid];
        o[((size_t)(b * SEQ + qi)) * DMODEL + h * HDIM + tid] = a * inv;
    }
}

// ============================ host side ====================================
static void launch_gemm(const float* A, const float* W, const float* bias,
                        const float* scale, float scaleC, const float* bias2,
                        const float* residual, float* C,
                        int M, int N, int K, int act, hipStream_t s)
{
    dim3 grid((N + TN - 1) / TN, (M + TM - 1) / TM);
    gemm_wmma_kernel<<<grid, 128, 0, s>>>(A, W, bias, scale, scaleC, bias2, residual,
                                          C, M, N, K, act);
}

extern "C" void kernel_launch(void* const* d_in, const int* in_sizes, int n_in,
                              void* d_out, int out_size, void* d_ws, size_t ws_size,
                              hipStream_t stream)
{
    const float* pts      = (const float*)d_in[0];
    const float* conv1_w  = (const float*)d_in[1];
    const float* conv1_b  = (const float*)d_in[2];
    const float* bn1_g    = (const float*)d_in[3];
    const float* bn1_b    = (const float*)d_in[4];
    const float* conv2_w  = (const float*)d_in[5];
    const float* conv2_b  = (const float*)d_in[6];
    const float* conv3_w  = (const float*)d_in[7];
    const float* conv3_b  = (const float*)d_in[8];
    const float* bn2_g    = (const float*)d_in[9];
    const float* bn2_b    = (const float*)d_in[10];
    const float* conv4_w  = (const float*)d_in[11];
    const float* conv4_b  = (const float*)d_in[12];
    const float* reduce_w = (const float*)d_in[13];
    const float* reduce_b = (const float*)d_in[14];
    const float* cls_tok  = (const float*)d_in[15];
    const float* cls_pos  = (const float*)d_in[16];
    const float* pos1_w   = (const float*)d_in[17];
    const float* pos1_b   = (const float*)d_in[18];
    const float* pos2_w   = (const float*)d_in[19];
    const float* pos2_b   = (const float*)d_in[20];
    const float* ln1_g    = (const float*)d_in[21];
    const float* ln1_b    = (const float*)d_in[22];
    const float* qkv_w    = (const float*)d_in[23];
    const float* qkv_b    = (const float*)d_in[24];
    const float* proj_w   = (const float*)d_in[25];
    const float* proj_b   = (const float*)d_in[26];
    const float* ln2_g    = (const float*)d_in[27];
    const float* ln2_b    = (const float*)d_in[28];
    const float* fc1_w    = (const float*)d_in[29];
    const float* fc1_b    = (const float*)d_in[30];
    const float* fc2_w    = (const float*)d_in[31];
    const float* fc2_b    = (const float*)d_in[32];
    (void)in_sizes; (void)n_in; (void)out_size; (void)ws_size;

    // ---- workspace layout (floats); ~295 MB total ----
    float* wsf = (float*)d_ws;
    size_t off = 0;
    auto alloc = [&](size_t n) { float* p = wsf + off; off += n; return p; };

    const int TOK  = BATCH * NGRP;        // 1024
    const int ROWS = BATCH * SEQ;         // 1026
    float* center = alloc((size_t)TOK * 3);
    int*   idx    = (int*)alloc((size_t)TOK * GS);
    float* tokens = alloc((size_t)TOK * DMODEL);
    float* posT   = alloc((size_t)TOK * DMODEL);
    float* x      = alloc((size_t)ROWS * DMODEL);
    float* pos    = alloc((size_t)ROWS * DMODEL);
    float* xi     = alloc((size_t)ROWS * DMODEL);
    float* hbuf   = alloc((size_t)ROWS * DMODEL);
    float* qkvb   = alloc((size_t)ROWS * 3 * DMODEL);
    float* attno  = alloc((size_t)ROWS * DMODEL);
    float* h2     = alloc((size_t)ROWS * DMODEL);
    float* fc1o   = alloc((size_t)ROWS * 4 * DMODEL);
    float* feat   = alloc((size_t)TOK * 256);
    float* posh   = alloc((size_t)TOK * 128);
    float* f0     = alloc((size_t)BATCH * NPTS * 6);
    float* bufA   = alloc((size_t)BATCH * NPTS * 512);   // ping
    float* bufB   = alloc((size_t)BATCH * NPTS * 512);   // pong
    float* knnscr = bufA;   // aliases bufA = 1024*32768 floats exactly

    const float bn_s1 = 1.0f / sqrtf(1.0f + 1e-5f);
    const int NPOINT_ALL = BATCH * NPTS;                 // 65536 rows in group MLP

    // ---- geometry ----
    fps_kernel<<<BATCH, 1024, 0, stream>>>(pts, center);
    knn_kernel<<<TOK, 256, 0, stream>>>(pts, center, idx, knnscr);
    {
        const int total = TOK * GS * 6;
        build_f0_kernel<<<(total + 255) / 256, 256, 0, stream>>>(pts, center, idx, f0);
    }

    // ---- group mini-PointNet (all WMMA GEMMs) ----
    launch_gemm(f0,   conv1_w, conv1_b, bn1_g, bn_s1, bn1_b, nullptr, bufA,
                NPOINT_ALL, 128, 6, 1, stream);                       // relu
    launch_gemm(bufA, conv2_w, conv2_b, nullptr, 1.0f, nullptr, nullptr, bufB,
                NPOINT_ALL, 256, 128, 0, stream);
    concat_max_kernel<<<TOK, 256, 0, stream>>>(bufB, bufA);           // -> 512ch
    launch_gemm(bufA, conv3_w, conv3_b, bn2_g, bn_s1, bn2_b, nullptr, bufB,
                NPOINT_ALL, 512, 512, 1, stream);                     // relu
    launch_gemm(bufB, conv4_w, conv4_b, nullptr, 1.0f, nullptr, nullptr, bufA,
                NPOINT_ALL, 256, 512, 0, stream);
    featmax_kernel<<<TOK, 256, 0, stream>>>(bufA, feat);

    // ---- tokens + positional embedding ----
    launch_gemm(feat, reduce_w, reduce_b, nullptr, 1.0f, nullptr, nullptr, tokens,
                TOK, DMODEL, 256, 0, stream);
    launch_gemm(center, pos1_w, pos1_b, nullptr, 1.0f, nullptr, nullptr, posh,
                TOK, 128, 3, 2, stream);                              // gelu
    launch_gemm(posh, pos2_w, pos2_b, nullptr, 1.0f, nullptr, nullptr, posT,
                TOK, DMODEL, 128, 0, stream);
    {
        const int total = ROWS * DMODEL;
        assemble_kernel<<<(total + 255) / 256, 256, 0, stream>>>(tokens, cls_tok, x);
        assemble_kernel<<<(total + 255) / 256, 256, 0, stream>>>(posT, cls_pos, pos);
    }

    // ---- transformer ----
    for (int i = 0; i < NLAYER; ++i) {
        ln_kernel<<<ROWS, 128, 0, stream>>>(x, pos, xi,
                                            ln1_g + (size_t)i * DMODEL,
                                            ln1_b + (size_t)i * DMODEL, hbuf);
        launch_gemm(hbuf, qkv_w + (size_t)i * 3 * DMODEL * DMODEL,
                    qkv_b + (size_t)i * 3 * DMODEL, nullptr, 1.0f, nullptr, nullptr,
                    qkvb, ROWS, 3 * DMODEL, DMODEL, 0, stream);
        attn_kernel<<<dim3(SEQ, NHEAD, BATCH), 128, 0, stream>>>(qkvb, attno);
        launch_gemm(attno, proj_w + (size_t)i * DMODEL * DMODEL,
                    proj_b + (size_t)i * DMODEL, nullptr, 1.0f, nullptr, xi,
                    x, ROWS, DMODEL, DMODEL, 0, stream);              // x = xi + proj(o)
        ln_kernel<<<ROWS, 128, 0, stream>>>(x, nullptr, nullptr,
                                            ln2_g + (size_t)i * DMODEL,
                                            ln2_b + (size_t)i * DMODEL, h2);
        launch_gemm(h2, fc1_w + (size_t)i * 4 * DMODEL * DMODEL,
                    fc1_b + (size_t)i * 4 * DMODEL, nullptr, 1.0f, nullptr, nullptr,
                    fc1o, ROWS, 4 * DMODEL, DMODEL, 2, stream);       // gelu
        float* dst = (i == NLAYER - 1) ? (float*)d_out : x;
        launch_gemm(fc1o, fc2_w + (size_t)i * DMODEL * 4 * DMODEL,
                    fc2_b + (size_t)i * DMODEL, nullptr, 1.0f, nullptr, x,
                    dst, ROWS, DMODEL, 4 * DMODEL, 0, stream);        // x += mlp
    }
}
